// MultiHeadAttention_10591389352069
// MI455X (gfx1250) — matrix operand order repaired and checked by reading, not compiled, not run on hardware
//
#include <hip/hip_runtime.h>
#include <hip/hip_bf16.h>

#define SEQ   3072
#define EDIM  1024
#define HEADS 16
#define HDIM  64
#define ROWS  16          // rows per fused workgroup (one WMMA M-tile)

typedef __attribute__((ext_vector_type(16))) _Float16 v16h;
typedef __attribute__((ext_vector_type(8)))  _Float16 v8h;
typedef __attribute__((ext_vector_type(8)))  float    v8f;

__device__ __forceinline__ v16h cat8h(v8h lo, v8h hi) {
  return __builtin_shufflevector(lo, hi, 0,1,2,3,4,5,6,7,8,9,10,11,12,13,14,15);
}
// A operand (16x32 f16): lane covers two contiguous 8-elem runs at p and p+16
__device__ __forceinline__ v16h load_a_f32(const float* p) {
  v8f lo = *(const v8f*)p;
  v8f hi = *(const v8f*)(p + 16);
  return cat8h(__builtin_convertvector(lo, v8h), __builtin_convertvector(hi, v8h));
}
__device__ __forceinline__ v16h load_a_f16(const _Float16* p) {
  return cat8h(*(const v8h*)p, *(const v8h*)(p + 16));
}
// B operand (32x16): lane covers two contiguous 8-elem runs at p and p+16
__device__ __forceinline__ v16h load_b_f32(const float* p) {
  v8f lo = *(const v8f*)p;
  v8f hi = *(const v8f*)(p + 16);
  return cat8h(__builtin_convertvector(lo, v8h), __builtin_convertvector(hi, v8h));
}
__device__ __forceinline__ v16h load_b_f16(const _Float16* p) {
  return cat8h(*(const v8h*)p, *(const v8h*)(p + 16));
}
__device__ __forceinline__ v8f wmma16(v16h a, v16h b, v8f c) {
  return __builtin_amdgcn_wmma_f32_16x16x32_f16(false, a, false, b, (short)0, c, false, false);
}

// ---------------------------------------------------------------------------
// Projection: Y[n][o] = sum_k X[n][k] * W[o][k]
// transposed==0 -> out f16 row-major [SEQ][EDIM]; transposed==1 -> out f16 [EDIM][SEQ]
// ---------------------------------------------------------------------------
__global__ __launch_bounds__(256)
void proj_kernel(const float* __restrict__ X, const float* __restrict__ W,
                 _Float16* __restrict__ out, int transposed)
{
  int wave = (int)((blockIdx.x * blockDim.x + threadIdx.x) >> 5);
  int lane = (int)(threadIdx.x & 31);
  int rt = wave / (EDIM / 64);
  int cg = wave % (EDIM / 64);
  int row0 = rt * 16, col0 = cg * 64;
  int m = lane & 15, half = lane >> 4;

  v8f acc[4] = {};
  const float* arow = X + (size_t)(row0 + m) * EDIM;
  for (int k0 = 0; k0 < EDIM; k0 += 32) {
    v16h a = load_a_f32(arow + k0 + half * 8);
#pragma unroll
    for (int t = 0; t < 4; ++t) {
      const float* brow = W + (size_t)(col0 + t * 16 + m) * EDIM + k0 + half * 8;
      acc[t] = wmma16(a, load_b_f32(brow), acc[t]);
    }
  }
#pragma unroll
  for (int t = 0; t < 4; ++t) {
    int col = col0 + t * 16 + m;
#pragma unroll
    for (int r = 0; r < 8; ++r) {
      int row = row0 + r + 8 * half;
      _Float16 v = (_Float16)acc[t][r];
      if (transposed) out[(size_t)col * SEQ + row] = v;
      else            out[(size_t)row * EDIM + col] = v;
    }
  }
}

// ---------------------------------------------------------------------------
// Fused attention for one (head, 16-row tile):
//   scores (WMMA) -> mask/scale -> LDS -> softmax in LDS -> single coalesced
//   attn write -> context (WMMA, K split over 8 waves, ds_add_f32 reduce).
// LDS: 192KB scores + 4KB ctx reduction (CDNA5 WGP has 320KB).
// ---------------------------------------------------------------------------
__global__ __launch_bounds__(256)
void attn_fused_kernel(const _Float16* __restrict__ Q16, const _Float16* __restrict__ K16,
                       const _Float16* __restrict__ Vt, const unsigned char* __restrict__ mask,
                       float* __restrict__ attn, float* __restrict__ ctx)
{
  __shared__ float sS[ROWS * SEQ];    // 192 KB: scores, then probabilities
  __shared__ float sC[ROWS * HDIM];   // 4 KB: context cross-wave reduction
  __shared__ float sInv[ROWS];

  int tid  = (int)threadIdx.x;
  int wave = tid >> 5, lane = tid & 31;
  int m = lane & 15, half = lane >> 4;
  int h    = (int)blockIdx.x / (SEQ / ROWS);
  int rt   = (int)blockIdx.x % (SEQ / ROWS);
  int row0 = rt * ROWS;

  for (int i = tid; i < ROWS * HDIM; i += 256) sC[i] = 0.0f;

  // ---- stage 1: scores = QK^T/8 with mask, into LDS. Q tile loaded once.
  const _Float16* arow = Q16 + (size_t)(row0 + m) * EDIM + h * HDIM;
  v16h aq0 = load_a_f16(arow + 0  + half * 8);
  v16h aq1 = load_a_f16(arow + 32 + half * 8);
  for (int cg = wave * 6; cg < wave * 6 + 6; ++cg) {      // 8 waves x 6 = 48 col groups
    int col0 = cg * 64;
    v8f acc[4] = {};
#pragma unroll
    for (int t = 0; t < 4; ++t) {
      const _Float16* brow = K16 + (size_t)(col0 + t * 16 + m) * EDIM + h * HDIM;
      acc[t] = wmma16(aq0, load_b_f16(brow + half * 8), acc[t]);
      acc[t] = wmma16(aq1, load_b_f16(brow + 32 + half * 8), acc[t]);
    }
#pragma unroll
    for (int t = 0; t < 4; ++t) {
      int col = col0 + t * 16 + m;
#pragma unroll
      for (int r = 0; r < 8; ++r) {
        int row = r + 8 * half;
        float s = acc[t][r] * 0.125f;                      // 1/sqrt(64)
        if (mask[(size_t)(row0 + row) * SEQ + col]) s = -1e9f;
        sS[row * SEQ + col] = s;
      }
    }
  }
  __syncthreads();

  // ---- stage 2: softmax per row; 16-thread teams, shuffle reductions
  int trow = tid >> 4;          // 0..15
  int c0   = tid & 15;
  float mx = -3.402823466e38f;
  for (int i = 0; i < SEQ / 16; ++i) mx = fmaxf(mx, sS[trow * SEQ + c0 + i * 16]);
#pragma unroll
  for (int off = 8; off > 0; off >>= 1) mx = fmaxf(mx, __shfl_xor(mx, off, 16));
  float sum = 0.0f;
  for (int i = 0; i < SEQ / 16; ++i) {
    float e = __expf(sS[trow * SEQ + c0 + i * 16] - mx);
    sS[trow * SEQ + c0 + i * 16] = e;
    sum += e;
  }
#pragma unroll
  for (int off = 8; off > 0; off >>= 1) sum += __shfl_xor(sum, off, 16);
  if (c0 == 0) sInv[trow] = 1.0f / sum;
  __syncthreads();

  // ---- stage 2c: normalize in LDS + the single coalesced global attn write
  float* gout = attn + (size_t)h * SEQ * SEQ + (size_t)row0 * SEQ;
  for (int idx = tid; idx < ROWS * SEQ; idx += 256) {
    float v = sS[idx] * sInv[idx / SEQ];
    sS[idx] = v;
    gout[idx] = v;
  }
  __syncthreads();

  // ---- stage 3: context = attn x V_h ; K-dim split across 8 waves
  v8f cacc[4] = {};
  int kbeg = wave * (SEQ / 8);
  for (int k0 = kbeg; k0 < kbeg + SEQ / 8; k0 += 32) {
    const float* ap = &sS[m * SEQ + k0 + half * 8];
    v8f lo = *(const v8f*)ap;
    v8f hi = *(const v8f*)(ap + 16);
    v16h a = cat8h(__builtin_convertvector(lo, v8h), __builtin_convertvector(hi, v8h));
#pragma unroll
    for (int t = 0; t < 4; ++t) {
      const _Float16* brow = Vt + (size_t)(h * HDIM + t * 16 + m) * SEQ + k0 + half * 8;
      cacc[t] = wmma16(a, load_b_f16(brow), cacc[t]);
    }
  }
#pragma unroll
  for (int t = 0; t < 4; ++t)
#pragma unroll
    for (int r = 0; r < 8; ++r)
      atomicAdd(&sC[(r + 8 * half) * HDIM + t * 16 + m], cacc[t][r]);
  __syncthreads();

  for (int i = tid; i < ROWS * HDIM; i += 256) {
    int row = i / HDIM, d = i % HDIM;
    ctx[(size_t)(row0 + row) * EDIM + h * HDIM + d] = sC[i];
  }
}

extern "C" void kernel_launch(void* const* d_in, const int* in_sizes, int n_in,
                              void* d_out, int out_size, void* d_ws, size_t ws_size,
                              hipStream_t stream)
{
  (void)in_sizes; (void)n_in; (void)out_size; (void)ws_size;
  const float*         inQ  = (const float*)d_in[0];
  const float*         inK  = (const float*)d_in[1];
  const float*         inV  = (const float*)d_in[2];
  const unsigned char* mask = (const unsigned char*)d_in[3];   // jnp bool = 1 byte
  const float*         Wq   = (const float*)d_in[4];
  const float*         Wv   = (const float*)d_in[5];

  float* ctx  = (float*)d_out;                       // [SEQ][EDIM]
  float* attn = ctx + (size_t)SEQ * EDIM;            // [HEADS][SEQ][SEQ]

  _Float16* Q16 = (_Float16*)d_ws;                   // [SEQ][EDIM]
  _Float16* K16 = Q16 + (size_t)SEQ * EDIM;          // [SEQ][EDIM]
  _Float16* Vt  = K16 + (size_t)SEQ * EDIM;          // [EDIM][SEQ]

  dim3 blk(256);
  const int projBlocks  = (SEQ / 16) * (EDIM / 64) * 32 / 256;   // 384
  const int fusedBlocks = HEADS * (SEQ / ROWS);                  // 3072

  proj_kernel<<<projBlocks, blk, 0, stream>>>(inQ, Wq, Q16, 0);
  proj_kernel<<<projBlocks, blk, 0, stream>>>(inK, Wq, K16, 0);
  proj_kernel<<<projBlocks, blk, 0, stream>>>(inV, Wv, Vt, 1);
  attn_fused_kernel<<<fusedBlocks, blk, 0, stream>>>(Q16, K16, Vt, mask, attn, ctx);
}
